// GATEncoder_1151051235460
// MI455X (gfx1250) — matrix operand order repaired and checked by reading, not compiled, not run on hardware
//
#include <hip/hip_runtime.h>
#include <hip/hip_bf16.h>
#include <math.h>

// ---------------------------------------------------------------------------
// Types for CDNA5 WMMA (wave32)
// ---------------------------------------------------------------------------
typedef __attribute__((ext_vector_type(16))) __bf16 bf16x16;
typedef __attribute__((ext_vector_type(8)))  __bf16 bf16x8;
typedef __attribute__((ext_vector_type(4)))  __bf16 bf16x4;
typedef __attribute__((ext_vector_type(8)))  float  f32x8;
typedef __attribute__((ext_vector_type(4)))  float  f32x4;

#define TILE_M 128
#define TILE_N 128
#define TILE_K 32
#define LDA 40   // padded LDS row stride (elements) for A tile [m][k]
#define LDB 40   // padded LDS row stride (elements) for B tile [n][k] (transposed)

// ---------------------------------------------------------------------------
// bf16 WMMA GEMM: C[M,N] = A[M,K] * B[K,N], all row-major f32 in global.
// f32 -> bf16 convert while staging to LDS; f32 accumulate.
// Block: 256 threads = 8 waves, wave grid 4(M) x 2(N); wave tile 32x64 via
// 2x4 v_wmma_f32_16x16x32_bf16. Software-pipelined global->reg->LDS staging,
// branch-free (address clamp + select) so loads issue in bulk.
// ---------------------------------------------------------------------------
__global__ __launch_bounds__(256)
void gat_gemm_bf16_wmma(const float* __restrict__ A, const float* __restrict__ B,
                        float* __restrict__ C, int M, int N, int K)
{
    __shared__ __align__(16) __bf16 As[TILE_M * LDA];
    __shared__ __align__(16) __bf16 Bs[TILE_N * LDB];

    const int tid  = threadIdx.x;
    const int lane = tid & 31;
    const int wid  = tid >> 5;
    const int wm   = wid & 3;   // M subtile (32 rows)
    const int wn   = wid >> 2;  // N subtile (64 cols)
    const int lm   = lane & 15;
    const int half = lane >> 4;

    const int blockM = blockIdx.y * TILE_M;
    const int blockN = blockIdx.x * TILE_N;

    // staging patterns
    const int a_r0 = tid >> 3;         // 0..31, rows r0 + 32*i
    const int a_c0 = (tid & 7) * 4;    // 0..28
    const int b_k0 = tid >> 5;         // 0..7, k rows b_k0 + 8*i
    const int b_n0 = (tid & 31) * 4;   // 0..124

    // clamped row addresses (OOB rows/cols only feed unstored C entries)
    int a_row[4], b_colbase;
    #pragma unroll
    for (int i = 0; i < 4; ++i) {
        int gr = blockM + a_r0 + 32 * i;
        a_row[i] = gr < M ? gr : M - 1;
    }
    {
        int gn = blockN + b_n0;
        b_colbase = (gn + 3 < N) ? gn : (N - 4);
    }

    f32x8 acc[2][4] = {};
    f32x4 ra[4], rb[4];

    const int KF   = K / TILE_K;        // full K tiles
    const int ktail = K - KF * TILE_K;

    auto load_full = [&](int k0) {
        #pragma unroll
        for (int i = 0; i < 4; ++i)
            ra[i] = *(const f32x4*)(A + (size_t)a_row[i] * K + k0 + a_c0);
        #pragma unroll
        for (int i = 0; i < 4; ++i)
            rb[i] = *(const f32x4*)(B + (size_t)(k0 + b_k0 + 8 * i) * N + b_colbase);
    };
    auto store_stage = [&]() {
        #pragma unroll
        for (int i = 0; i < 4; ++i) {
            bf16x4 v;
            #pragma unroll
            for (int j = 0; j < 4; ++j) v[j] = (__bf16)ra[i][j];
            *(bf16x4*)(&As[(a_r0 + 32 * i) * LDA + a_c0]) = v;
        }
        #pragma unroll
        for (int i = 0; i < 4; ++i)
            #pragma unroll
            for (int j = 0; j < 4; ++j)
                Bs[(b_n0 + j) * LDB + b_k0 + 8 * i] = (__bf16)rb[i][j];
    };
    auto compute_tile = [&]() {
        bf16x16 af[2], bfr[4];
        #pragma unroll
        for (int mi = 0; mi < 2; ++mi) {
            const __bf16* p = &As[(wm * 32 + mi * 16 + lm) * LDA];
            bf16x8 lo = *(const bf16x8*)(p + half * 8);
            bf16x8 hi = *(const bf16x8*)(p + 16 + half * 8);
            af[mi] = __builtin_shufflevector(lo, hi,
                0,1,2,3,4,5,6,7,8,9,10,11,12,13,14,15);
        }
        #pragma unroll
        for (int ni = 0; ni < 4; ++ni) {
            const __bf16* p = &Bs[(wn * 64 + ni * 16 + lm) * LDB + half * 8];
            bf16x8 lo = *(const bf16x8*)(p);
            bf16x8 hi = *(const bf16x8*)(p + 16);
            bfr[ni] = __builtin_shufflevector(lo, hi,
                0,1,2,3,4,5,6,7,8,9,10,11,12,13,14,15);
        }
        #pragma unroll
        for (int mi = 0; mi < 2; ++mi)
            #pragma unroll
            for (int ni = 0; ni < 4; ++ni)
                acc[mi][ni] = __builtin_amdgcn_wmma_f32_16x16x32_bf16(
                    false, af[mi], false, bfr[ni],
                    (short)0, acc[mi][ni], false, false);
    };

    if (KF > 0) {
        load_full(0);
        store_stage();
        __syncthreads();
        for (int t = 0; t < KF; ++t) {
            bool has_next = (t + 1) < KF;
            if (has_next) load_full((t + 1) * TILE_K);   // issue early, wait later
            compute_tile();
            __syncthreads();                              // readers done
            if (has_next) store_stage();
            __syncthreads();                              // writes visible
        }
    }

    if (ktail > 0) {
        int k0 = KF * TILE_K;
        // A tail: clamped scalar loads + zero-select (branch-free)
        #pragma unroll
        for (int i = 0; i < 4; ++i) {
            #pragma unroll
            for (int j = 0; j < 4; ++j) {
                int gc = k0 + a_c0 + j;
                int cc = gc < K ? gc : K - 1;
                float v = A[(size_t)a_row[i] * K + cc];
                As[(a_r0 + 32 * i) * LDA + a_c0 + j] = (__bf16)(gc < K ? v : 0.0f);
            }
        }
        // B tail: clamped-row vector load + zero-select
        #pragma unroll
        for (int i = 0; i < 4; ++i) {
            int gk = k0 + b_k0 + 8 * i;
            int kk = gk < K ? gk : K - 1;
            f32x4 v = *(const f32x4*)(B + (size_t)kk * N + b_colbase);
            float z = (gk < K) ? 1.0f : 0.0f;
            #pragma unroll
            for (int j = 0; j < 4; ++j)
                Bs[(b_n0 + j) * LDB + b_k0 + 8 * i] = (__bf16)(v[j] * z);
        }
        __syncthreads();
        compute_tile();
    }

    // writeback: VGPR r -> m = r + 8*half, n = lm
    #pragma unroll
    for (int mi = 0; mi < 2; ++mi)
        #pragma unroll
        for (int ni = 0; ni < 4; ++ni)
            #pragma unroll
            for (int r = 0; r < 8; ++r) {
                int gr = blockM + wm * 32 + mi * 16 + half * 8 + r;
                int gc = blockN + wn * 64 + ni * 16 + lm;
                if (gr < M && gc < N) C[(size_t)gr * N + gc] = acc[mi][ni][r];
            }
}

// ---------------------------------------------------------------------------
// Helpers: ordered-int encoding of float for atomicMax-based segment max
// ---------------------------------------------------------------------------
__device__ __forceinline__ int enc_f(float f) {
    int i = __float_as_int(f);
    return i >= 0 ? i : ~i;
}
__device__ __forceinline__ float dec_f(int i) {
    return __int_as_float(i >= 0 ? i : ~i);
}

__global__ void fill_f32(float* p, float v, size_t n) {
    size_t t = (size_t)blockIdx.x * blockDim.x + threadIdx.x;
    if (t < n) p[t] = v;
}
__global__ void fill_i32(int* p, int v, size_t n) {
    size_t t = (size_t)blockIdx.x * blockDim.x + threadIdx.x;
    if (t < n) p[t] = v;
}

// ---------------------------------------------------------------------------
// alpha_s[n,h] = dot(h[n,h,:], a_src[h,:]); alpha_d likewise. One wave per
// (node, head), wave32 shuffle reduction.
// ---------------------------------------------------------------------------
__global__ void gat_alpha(const float* __restrict__ h,
                          const float* __restrict__ a_src,
                          const float* __restrict__ a_dst,
                          float* __restrict__ as, float* __restrict__ ad,
                          int Nn, int H, int C)
{
    int wv   = (int)(((size_t)blockIdx.x * blockDim.x + threadIdx.x) >> 5);
    int lane = threadIdx.x & 31;
    if (wv >= Nn * H) return;
    int n  = wv / H;
    int hh = wv % H;
    const float* hp  = h + (size_t)n * H * C + (size_t)hh * C;
    const float* asp = a_src + (size_t)hh * C;
    const float* adp = a_dst + (size_t)hh * C;
    float s = 0.0f, d = 0.0f;
    for (int c = lane; c < C; c += 32) {
        float v = hp[c];
        s += v * asp[c];
        d += v * adp[c];
    }
#pragma unroll
    for (int off = 16; off > 0; off >>= 1) {
        s += __shfl_xor(s, off, 32);
        d += __shfl_xor(d, off, 32);
    }
    if (lane == 0) { as[wv] = s; ad[wv] = d; }
}

// edge list = concat(edge_index, self-loops). edge_index is int64 [2,E].
__device__ __forceinline__ void edge_sd(const long long* __restrict__ eidx,
                                        int E, int e, long long& sv, long long& dv)
{
    if (e < E) { sv = eidx[e]; dv = eidx[(size_t)E + e]; }
    else       { sv = dv = (long long)(e - E); }
}

// ---------------------------------------------------------------------------
// Phase 1: m[dst,h] = max over incoming edges of leaky_relu(as[src]+ad[dst])
// ---------------------------------------------------------------------------
__global__ void gat_edge_max(const long long* __restrict__ eidx, int E, int Nn,
                             const float* __restrict__ as, const float* __restrict__ ad,
                             int* __restrict__ m, int H)
{
    int t  = blockIdx.x * blockDim.x + threadIdx.x;
    int Et = E + Nn;
    if (t >= Et * H) return;
    int e  = t / H;
    int hh = t - e * H;
    long long sv, dv;
    edge_sd(eidx, E, e, sv, dv);
    float v = as[(size_t)sv * H + hh] + ad[(size_t)dv * H + hh];
    v = v > 0.0f ? v : 0.2f * v;  // leaky_relu, slope 0.2
    atomicMax(&m[(size_t)dv * H + hh], enc_f(v));
}

// ---------------------------------------------------------------------------
// Phase 2: ex[e,h] = exp(e_val - m[dst,h]); den[dst,h] += ex
// ---------------------------------------------------------------------------
__global__ void gat_edge_exp(const long long* __restrict__ eidx, int E, int Nn,
                             const float* __restrict__ as, const float* __restrict__ ad,
                             const int* __restrict__ m, float* __restrict__ ex,
                             float* __restrict__ den, int H)
{
    int t  = blockIdx.x * blockDim.x + threadIdx.x;
    int Et = E + Nn;
    if (t >= Et * H) return;
    int e  = t / H;
    int hh = t - e * H;
    long long sv, dv;
    edge_sd(eidx, E, e, sv, dv);
    float v = as[(size_t)sv * H + hh] + ad[(size_t)dv * H + hh];
    v = v > 0.0f ? v : 0.2f * v;
    float x = __expf(v - dec_f(m[(size_t)dv * H + hh]));
    ex[(size_t)e * H + hh] = x;
    atomicAdd(&den[(size_t)dv * H + hh], x);
}

// ---------------------------------------------------------------------------
// Phase 3: out[dst,h,:] += h[src,h,:] * ex[e,h]/(den[dst,h]+1e-16)
// One block per edge; float atomics into out. C = 1<<cshift.
// ---------------------------------------------------------------------------
__global__ __launch_bounds__(256)
void gat_edge_scatter(const long long* __restrict__ eidx, int E, int Nn,
                      const float* __restrict__ h, const float* __restrict__ ex,
                      const float* __restrict__ den, float* __restrict__ out,
                      int H, int cshift)
{
    int e = blockIdx.x;
    long long sv, dv;
    edge_sd(eidx, E, e, sv, dv);
    __shared__ float w[16];
    if (threadIdx.x < H) {
        float d = den[(size_t)dv * H + threadIdx.x];
        w[threadIdx.x] = ex[(size_t)e * H + threadIdx.x] / (d + 1e-16f);
    }
    __syncthreads();
    int HC = H << cshift;
    const float* hs = h   + (size_t)sv * HC;
    float*       od = out + (size_t)dv * HC;
    for (int i = threadIdx.x; i < HC; i += blockDim.x)
        atomicAdd(&od[i], hs[i] * w[i >> cshift]);
}

// out = relu(out + bias)  (in place), bias length HC broadcast over rows
__global__ void gat_finalize_concat(float* __restrict__ out,
                                    const float* __restrict__ bias,
                                    size_t total, int HC)
{
    size_t t = (size_t)blockIdx.x * blockDim.x + threadIdx.x;
    if (t >= total) return;
    float v = out[t] + bias[t % HC];
    out[t] = v > 0.0f ? v : 0.0f;
}

// y[n,c] = relu(mean_h(out[n,h,c]) + b[c])
__global__ void gat_finalize_mean(const float* __restrict__ out,
                                  const float* __restrict__ bias,
                                  float* __restrict__ y, int Nn, int H, int C)
{
    int t = blockIdx.x * blockDim.x + threadIdx.x;
    if (t >= Nn * C) return;
    int n = t / C;
    int c = t - n * C;
    float s = 0.0f;
    for (int hh = 0; hh < H; ++hh)
        s += out[(size_t)n * H * C + (size_t)hh * C + c];
    s = s / (float)H + bias[c];
    y[t] = s > 0.0f ? s : 0.0f;
}

// ---------------------------------------------------------------------------
// Host orchestration
// ---------------------------------------------------------------------------
static inline int cdiv(long long a, long long b) { return (int)((a + b - 1) / b); }

extern "C" void kernel_launch(void* const* d_in, const int* in_sizes, int n_in,
                              void* d_out, int out_size, void* d_ws, size_t ws_size,
                              hipStream_t stream)
{
    const float*     x      = (const float*)d_in[0];
    const long long* eidx   = (const long long*)d_in[1];  // int64 [2,E]
    const float*     W1     = (const float*)d_in[2];
    const float*     a_src1 = (const float*)d_in[3];
    const float*     a_dst1 = (const float*)d_in[4];
    const float*     b1     = (const float*)d_in[5];
    const float*     W2     = (const float*)d_in[6];
    const float*     a_src2 = (const float*)d_in[7];
    const float*     a_dst2 = (const float*)d_in[8];
    const float*     b2     = (const float*)d_in[9];
    float*           y      = (float*)d_out;

    const int H   = 10;
    const int HC1 = in_sizes[3];            // H*C1 = 1280
    const int C1  = HC1 / H;                // 128
    const int Fin = in_sizes[2] / HC1;      // 2000
    const int Nn  = in_sizes[0] / Fin;      // 10000
    const int E   = in_sizes[1] / 2;        // 150000
    const int HC2 = in_sizes[6] / HC1;      // 640
    const int LAT = HC2 / H;                // 64
    const int Et  = E + Nn;
    const int cs1 = __builtin_ctz(C1);      // 7
    const int cs2 = __builtin_ctz(LAT);     // 6

    // workspace carve-up (floats)
    float* ws   = (float*)d_ws;
    float* h1   = ws;                       // N*HC1
    float* out1 = h1   + (size_t)Nn * HC1;  // N*HC1 (becomes relu(h1_agg+b1))
    float* h2   = out1 + (size_t)Nn * HC1;  // N*HC2
    float* out2 = h2   + (size_t)Nn * HC2;  // N*HC2
    float* ex1  = out2 + (size_t)Nn * HC2;  // Et*H
    float* ex2  = ex1  + (size_t)Et * H;    // Et*H
    float* as1  = ex2  + (size_t)Et * H;    // N*H each below
    float* ad1  = as1  + (size_t)Nn * H;
    float* den1 = ad1  + (size_t)Nn * H;
    float* as2  = den1 + (size_t)Nn * H;
    float* ad2  = as2  + (size_t)Nn * H;
    float* den2 = ad2  + (size_t)Nn * H;
    int*   m1   = (int*)(den2 + (size_t)Nn * H);  // N*H
    int*   m2   = m1   + (size_t)Nn * H;

    const int NEG_INF_ENC = 0x007FFFFF;  // enc_f(-inf)
    const int TB = 256;

    // ---- init accumulators
    fill_i32<<<cdiv((long long)Nn * H, TB), TB, 0, stream>>>(m1, NEG_INF_ENC, (size_t)Nn * H);
    fill_i32<<<cdiv((long long)Nn * H, TB), TB, 0, stream>>>(m2, NEG_INF_ENC, (size_t)Nn * H);
    fill_f32<<<cdiv((long long)Nn * H, TB), TB, 0, stream>>>(den1, 0.0f, (size_t)Nn * H);
    fill_f32<<<cdiv((long long)Nn * H, TB), TB, 0, stream>>>(den2, 0.0f, (size_t)Nn * H);
    fill_f32<<<cdiv((long long)Nn * HC1, TB), TB, 0, stream>>>(out1, 0.0f, (size_t)Nn * HC1);
    fill_f32<<<cdiv((long long)Nn * HC2, TB), TB, 0, stream>>>(out2, 0.0f, (size_t)Nn * HC2);

    // ================= layer 1 =================
    {
        dim3 grid(cdiv(HC1, TILE_N), cdiv(Nn, TILE_M));
        gat_gemm_bf16_wmma<<<grid, TB, 0, stream>>>(x, W1, h1, Nn, HC1, Fin);
    }
    gat_alpha<<<cdiv((long long)Nn * H * 32, TB), TB, 0, stream>>>(
        h1, a_src1, a_dst1, as1, ad1, Nn, H, C1);
    gat_edge_max<<<cdiv((long long)Et * H, TB), TB, 0, stream>>>(
        eidx, E, Nn, as1, ad1, m1, H);
    gat_edge_exp<<<cdiv((long long)Et * H, TB), TB, 0, stream>>>(
        eidx, E, Nn, as1, ad1, m1, ex1, den1, H);
    gat_edge_scatter<<<Et, TB, 0, stream>>>(
        eidx, E, Nn, h1, ex1, den1, out1, H, cs1);
    gat_finalize_concat<<<cdiv((long long)Nn * HC1, TB), TB, 0, stream>>>(
        out1, b1, (size_t)Nn * HC1, HC1);

    // ================= layer 2 =================
    {
        dim3 grid(cdiv(HC2, TILE_N), cdiv(Nn, TILE_M));
        gat_gemm_bf16_wmma<<<grid, TB, 0, stream>>>(out1, W2, h2, Nn, HC2, HC1);
    }
    gat_alpha<<<cdiv((long long)Nn * H * 32, TB), TB, 0, stream>>>(
        h2, a_src2, a_dst2, as2, ad2, Nn, H, LAT);
    gat_edge_max<<<cdiv((long long)Et * H, TB), TB, 0, stream>>>(
        eidx, E, Nn, as2, ad2, m2, H);
    gat_edge_exp<<<cdiv((long long)Et * H, TB), TB, 0, stream>>>(
        eidx, E, Nn, as2, ad2, m2, ex2, den2, H);
    gat_edge_scatter<<<Et, TB, 0, stream>>>(
        eidx, E, Nn, h2, ex2, den2, out2, H, cs2);
    gat_finalize_mean<<<cdiv((long long)Nn * LAT, TB), TB, 0, stream>>>(
        out2, b2, y, Nn, H, LAT);
}
